// SynCo_17695265259639
// MI455X (gfx1250) — matrix operand order repaired and checked by reading, not compiled, not run on hardware
//
#include <hip/hip_runtime.h>
#include <stdint.h>

#define B_    1024
#define C_    256
#define K_    65536
#define NH_   1024
#define SUMN_ 576
#define LD_   (1 + K_ + SUMN_)
#define INV_TEMP 5.0f

typedef __bf16 v16bf  __attribute__((ext_vector_type(16)));
typedef __bf16 bf16x8 __attribute__((ext_vector_type(8)));
typedef float  v8f    __attribute__((ext_vector_type(8)));

// ---------------------------------------------------------------------------
// K1: normalize q and key_k rows, emit q_norm (f32 + bf16 hi/lo), l_pos/TEMP
// ---------------------------------------------------------------------------
__global__ void __launch_bounds__(256) k_norm_qk(
    const float* __restrict__ q, const float* __restrict__ kk,
    float* __restrict__ qn, __bf16* __restrict__ qhi, __bf16* __restrict__ qlo,
    float* __restrict__ out) {
  const int b = blockIdx.x, t = threadIdx.x;
  __shared__ float red[256];
  float qv = q[b * C_ + t];
  float kv = kk[b * C_ + t];
  auto reduceSum = [&](float v) -> float {
    red[t] = v; __syncthreads();
    for (int s = 128; s > 0; s >>= 1) { if (t < s) red[t] += red[t + s]; __syncthreads(); }
    float r = red[0]; __syncthreads(); return r;
  };
  float sq2 = reduceSum(qv * qv);
  float sk2 = reduceSum(kv * kv);
  float qni = 1.0f / fmaxf(sqrtf(sq2), 1e-12f);
  float kni = 1.0f / fmaxf(sqrtf(sk2), 1e-12f);
  float qvn = qv * qni, kvn = kv * kni;
  float dp = reduceSum(qvn * kvn);
  qn[b * C_ + t] = qvn;
  __bf16 h = (__bf16)qvn;
  qhi[b * C_ + t] = h;
  qlo[b * C_ + t] = (__bf16)(qvn - (float)h);
  if (t == 0) out[(size_t)b * LD_] = dp * INV_TEMP;
}

// ---------------------------------------------------------------------------
// K2a: per-column inverse L2 norm of queue [C,K] (coalesced over k)
// ---------------------------------------------------------------------------
__global__ void __launch_bounds__(256) k_colnorm(
    const float* __restrict__ queue, float* __restrict__ invn) {
  const int k = blockIdx.x * 256 + threadIdx.x;
  float s = 0.f;
  for (int c = 0; c < C_; ++c) { float v = queue[(size_t)c * K_ + k]; s += v * v; }
  invn[k] = 1.0f / fmaxf(sqrtf(s), 1e-12f);
}

// ---------------------------------------------------------------------------
// K2b: transpose + normalize + bf16 hi/lo split via LDS tile (coalesced both ways)
// Output: Qhi/Qlo as [K, C] row-major bf16
// ---------------------------------------------------------------------------
__global__ void __launch_bounds__(256) k_transpose_split(
    const float* __restrict__ queue, const float* __restrict__ invn,
    __bf16* __restrict__ Qhi, __bf16* __restrict__ Qlo) {
  __shared__ float tile[64][65];
  const int k0 = blockIdx.x * 64, c0 = blockIdx.y * 64;
  const int t = threadIdx.x;
  for (int i = 0; i < 16; ++i) {
    int lin = t + i * 256;
    int ci = lin >> 6, ki = lin & 63;
    tile[ci][ki] = queue[(size_t)(c0 + ci) * K_ + (k0 + ki)];
  }
  __syncthreads();
  for (int i = 0; i < 16; ++i) {
    int lin = t + i * 256;
    int ki = lin >> 6, ci = lin & 63;
    float v = tile[ci][ki] * invn[k0 + ki];
    __bf16 h = (__bf16)v;
    size_t o = (size_t)(k0 + ki) * C_ + (c0 + ci);
    Qhi[o] = h;
    Qlo[o] = (__bf16)(v - (float)h);
  }
}

// ---------------------------------------------------------------------------
// K3: l_neg GEMM with v_wmma_f32_16x16x32_bf16, split-bf16 3-term precision.
// Each wave: one 16(b) x 64(k) tile; block = 8 waves -> 16x512; grid (128,64).
// ---------------------------------------------------------------------------
static __device__ __forceinline__ v16bf cat8(bf16x8 a, bf16x8 b) {
  return __builtin_shufflevector(a, b, 0,1,2,3,4,5,6,7,8,9,10,11,12,13,14,15);
}

__global__ void __launch_bounds__(256) k_gemm(
    const __bf16* __restrict__ qhi, const __bf16* __restrict__ qlo,
    const __bf16* __restrict__ Qhi, const __bf16* __restrict__ Qlo,
    float* __restrict__ out) {
  const int lane = threadIdx.x & 31;
  const int wave = threadIdx.x >> 5;
  const int half = lane >> 4;
  const int m    = lane & 15;
  const int b0 = blockIdx.y * 16;
  const int k0 = (blockIdx.x * 8 + wave) * 64;

  const __bf16* qrow_h = qhi + (size_t)(b0 + m) * C_;
  const __bf16* qrow_l = qlo + (size_t)(b0 + m) * C_;

  v8f zero = {};
  v8f acc[4] = {zero, zero, zero, zero};

  for (int s = 0; s < 8; ++s) {
    // A tile (16x32 bf16): per-lane two contiguous 8-element chunks
    const int o0 = s * 32 + half * 8;
    const int o1 = s * 32 + 16 + half * 8;
    v16bf Ah = cat8(*(const bf16x8*)(qrow_h + o0), *(const bf16x8*)(qrow_h + o1));
    v16bf Al = cat8(*(const bf16x8*)(qrow_l + o0), *(const bf16x8*)(qrow_l + o1));

#pragma unroll
    for (int t4 = 0; t4 < 4; ++t4) {
      const size_t brow = (size_t)(k0 + t4 * 16 + m) * C_ + s * 32 + half * 8;
      const __bf16* bh = Qhi + brow;
      const __bf16* bl = Qlo + brow;
      if (s < 7) __builtin_prefetch(bh + 32, 0, 1);   // -> global_prefetch_b8
      v16bf Bh = cat8(*(const bf16x8*)(bh), *(const bf16x8*)(bh + 16));
      v16bf Bl = cat8(*(const bf16x8*)(bl), *(const bf16x8*)(bl + 16));
      // (Ah+Al)*(Bh+Bl) ~= Ah*Bh + Al*Bh + Ah*Bl  (drop lo*lo)
      acc[t4] = __builtin_amdgcn_wmma_f32_16x16x32_bf16(false, Ah, false, Bh,
                                                        (short)0, acc[t4], false, false);
      acc[t4] = __builtin_amdgcn_wmma_f32_16x16x32_bf16(false, Al, false, Bh,
                                                        (short)0, acc[t4], false, false);
      acc[t4] = __builtin_amdgcn_wmma_f32_16x16x32_bf16(false, Ah, false, Bl,
                                                        (short)0, acc[t4], false, false);
    }
  }

  // D layout: VGPR v -> row b0 + half*8 + v, col k0 + t4*16 + (lane&15)
#pragma unroll
  for (int t4 = 0; t4 < 4; ++t4) {
    const int col = k0 + t4 * 16 + m;
#pragma unroll
    for (int v = 0; v < 8; ++v) {
      const int row = b0 + half * 8 + v;
      out[(size_t)row * LD_ + 1 + col] = acc[t4][v] * INV_TEMP;
    }
  }
}

// ---------------------------------------------------------------------------
// K4: exact sorted top-1024 per row (matches jax.lax.top_k ordering).
// Histogram select on top-11 monotone-key bits, gather, LDS bitonic sorts.
// ---------------------------------------------------------------------------
__device__ __forceinline__ unsigned monokey(float x) {
  unsigned u = __float_as_uint(x);
  return (u & 0x80000000u) ? ~u : (u | 0x80000000u);
}

__global__ void __launch_bounds__(256) k_topk(
    const float* __restrict__ out, int* __restrict__ ih) {
  const int b = blockIdx.x, t = threadIdx.x;
  const float* row = out + (size_t)b * LD_ + 1;

  __shared__ unsigned hist[2048];
  __shared__ unsigned ckey[NH_];
  __shared__ int      cidx[NH_];
  __shared__ unsigned ekey[2048];
  __shared__ int      eidx[2048];
  __shared__ unsigned s_above, s_bsel, s_ncand, s_neq;

  for (int i = t; i < 2048; i += 256) hist[i] = 0u;
  __syncthreads();
  for (int i = t; i < K_; i += 256)
    atomicAdd(&hist[monokey(row[i]) >> 21], 1u);
  __syncthreads();

  if (t == 0) {
    unsigned cum = 0, bsel = 0;
    for (int bin = 2047; bin >= 0; --bin) {
      unsigned h = hist[bin];
      if (cum + h >= (unsigned)NH_) { bsel = (unsigned)bin; break; }
      cum += h;
    }
    s_above = cum; s_bsel = bsel; s_ncand = 0; s_neq = 0;
  }
  __syncthreads();
  const unsigned bsel = s_bsel;

  for (int i = t; i < K_; i += 256) {
    unsigned key = monokey(row[i]);
    unsigned top = key >> 21;
    if (top > bsel) {
      unsigned p = atomicAdd(&s_ncand, 1u);
      ckey[p] = key; cidx[p] = i;
    } else if (top == bsel) {
      unsigned p = atomicAdd(&s_neq, 1u);
      if (p < 2048u) { ekey[p] = key; eidx[p] = i; }
    }
  }
  __syncthreads();

  const unsigned neq = s_neq > 2048u ? 2048u : s_neq;
  for (int i = t; i < 2048; i += 256)
    if ((unsigned)i >= neq) { ekey[i] = 0u; eidx[i] = 0x7fffffff; }
  __syncthreads();

  // bitonic sort boundary-bin list (descending key, ascending idx on ties)
  for (int k = 2; k <= 2048; k <<= 1)
    for (int j = k >> 1; j > 0; j >>= 1) {
      for (int i = t; i < 2048; i += 256) {
        int l = i ^ j;
        if (l > i) {
          unsigned ki = ekey[i], kl = ekey[l];
          int ii = eidx[i], il = eidx[l];
          bool before = (ki > kl) || (ki == kl && ii < il);
          bool desc = ((i & k) == 0);
          if (desc ? !before : before) {
            ekey[i] = kl; ekey[l] = ki; eidx[i] = il; eidx[l] = ii;
          }
        }
      }
      __syncthreads();
    }

  const int above = (int)s_above;
  const int needed = NH_ - above;
  for (int i = t; i < needed; i += 256) { ckey[above + i] = ekey[i]; cidx[above + i] = eidx[i]; }
  __syncthreads();

  // bitonic sort full 1024-candidate list descending
  for (int k = 2; k <= NH_; k <<= 1)
    for (int j = k >> 1; j > 0; j >>= 1) {
      for (int i = t; i < NH_; i += 256) {
        int l = i ^ j;
        if (l > i) {
          unsigned ki = ckey[i], kl = ckey[l];
          int ii = cidx[i], il = cidx[l];
          bool before = (ki > kl) || (ki == kl && ii < il);
          bool desc = ((i & k) == 0);
          if (desc ? !before : before) {
            ckey[i] = kl; ckey[l] = ki; cidx[i] = il; cidx[l] = ii;
          }
        }
      }
      __syncthreads();
    }

  for (int i = t; i < NH_; i += 256) ih[b * NH_ + i] = cidx[i];
}

// ---------------------------------------------------------------------------
// K5: 576 synthetic hard negatives per row; one wave per synth row,
// lane handles 8 channels, wave32 butterfly reduction for norm & dot.
// ---------------------------------------------------------------------------
__global__ void __launch_bounds__(256) k_synth(
    const float* __restrict__ qn,
    const __bf16* __restrict__ Qhi, const __bf16* __restrict__ Qlo,
    const int* __restrict__ ih,
    const float* __restrict__ u1, const float* __restrict__ u2,
    const float* __restrict__ u3, const float* __restrict__ noise4,
    const int* __restrict__ idx1, const int* __restrict__ idx2,
    const int* __restrict__ idx3a, const int* __restrict__ idx3b,
    const int* __restrict__ idx4, const int* __restrict__ idx5,
    const int* __restrict__ idx6,
    float* __restrict__ out) {
  const int b = blockIdx.x, t = threadIdx.x;
  __shared__ float sq[C_];
  sq[t] = qn[b * C_ + t];
  __syncthreads();

  const int wave = t >> 5, lane = t & 31;
  const int* ihb = ih + b * NH_;

  for (int r = wave; r < SUMN_; r += 8) {
    int sel, sel2 = 0, mode;
    float ca = 0.f, cb = 0.f;
    const float* np = nullptr;
    if (r < 128)      { int j = r;       sel = ihb[idx1[b*128 + j]]; ca = u1[b*128 + j] * 0.5f; mode = 0; }
    else if (r < 256) { int j = r - 128; sel = ihb[idx2[b*128 + j]]; cb = 1.0f + u2[b*128 + j] * 0.5f; mode = 1; }
    else if (r < 384) { int j = r - 256; sel = ihb[idx3a[b*128 + j]]; sel2 = ihb[idx3b[b*128 + j]];
                        ca = u3[b*128 + j]; mode = 2; }
    else if (r < 448) { int j = r - 384; sel = ihb[idx4[b*64 + j]];
                        np = noise4 + ((size_t)b * 64 + j) * C_; mode = 3; }
    else if (r < 512) { int j = r - 448; sel = ihb[idx5[b*64 + j]]; mode = 4; }
    else              { int j = r - 512; sel = ihb[idx6[b*64 + j]]; mode = 5; }

    const int cbase = lane * 8;
    bf16x8 vh = *(const bf16x8*)(Qhi + (size_t)sel * C_ + cbase);
    bf16x8 vl = *(const bf16x8*)(Qlo + (size_t)sel * C_ + cbase);
    bf16x8 wh = vh, wl = vl;
    if (mode == 2) {
      wh = *(const bf16x8*)(Qhi + (size_t)sel2 * C_ + cbase);
      wl = *(const bf16x8*)(Qlo + (size_t)sel2 * C_ + cbase);
    }

    float ss = 0.f, dot = 0.f;
#pragma unroll
    for (int jj = 0; jj < 8; ++jj) {
      const int c = cbase + jj;
      const float n  = (float)vh[jj] + (float)vl[jj];
      const float qv = sq[c];
      float v;
      switch (mode) {
        case 0: v = ca * qv + (1.f - ca) * n; break;
        case 1: v = qv + cb * (n - qv); break;
        case 2: { float n2 = (float)wh[jj] + (float)wl[jj]; v = ca * n + (1.f - ca) * n2; } break;
        case 3: v = n + 0.1f * np[c]; break;
        case 4: v = n + 0.01f * qv; break;
        default: v = n + 0.01f * ((qv > 0.f) ? 1.f : (qv < 0.f ? -1.f : 0.f)); break;
      }
      ss += v * v; dot += v * qv;
    }
    for (int off = 16; off >= 1; off >>= 1) {   // wave32 butterfly
      ss  += __shfl_xor(ss,  off);
      dot += __shfl_xor(dot, off);
    }
    if (lane == 0)
      out[(size_t)b * LD_ + 1 + K_ + r] = dot / fmaxf(sqrtf(ss), 1e-12f) * INV_TEMP;
  }
}

// ---------------------------------------------------------------------------
extern "C" void kernel_launch(void* const* d_in, const int* in_sizes, int n_in,
                              void* d_out, int out_size, void* d_ws, size_t ws_size,
                              hipStream_t stream) {
  (void)in_sizes; (void)n_in; (void)out_size; (void)ws_size;
  const float* q      = (const float*)d_in[0];
  const float* kk     = (const float*)d_in[1];
  const float* queue  = (const float*)d_in[2];
  const float* u1     = (const float*)d_in[3];
  const float* u2     = (const float*)d_in[4];
  const float* u3     = (const float*)d_in[5];
  const float* noise4 = (const float*)d_in[6];
  const int*   idx1   = (const int*)d_in[7];
  const int*   idx2   = (const int*)d_in[8];
  const int*   idx3a  = (const int*)d_in[9];
  const int*   idx3b  = (const int*)d_in[10];
  const int*   idx4   = (const int*)d_in[11];
  const int*   idx5   = (const int*)d_in[12];
  const int*   idx6   = (const int*)d_in[13];
  float* out = (float*)d_out;

  char* ws = (char*)d_ws;
  size_t off = 0;
  auto alloc = [&](size_t bytes) -> void* {
    size_t a = (off + 255) & ~(size_t)255;
    off = a + bytes;
    return (void*)(ws + a);
  };
  float*  qn   = (float*) alloc((size_t)B_ * C_ * sizeof(float));
  __bf16* qhi  = (__bf16*)alloc((size_t)B_ * C_ * sizeof(__bf16));
  __bf16* qlo  = (__bf16*)alloc((size_t)B_ * C_ * sizeof(__bf16));
  float*  invc = (float*) alloc((size_t)K_ * sizeof(float));
  __bf16* Qhi  = (__bf16*)alloc((size_t)K_ * C_ * sizeof(__bf16));
  __bf16* Qlo  = (__bf16*)alloc((size_t)K_ * C_ * sizeof(__bf16));
  int*    ih   = (int*)   alloc((size_t)B_ * NH_ * sizeof(int));

  k_norm_qk<<<dim3(B_), dim3(256), 0, stream>>>(q, kk, qn, qhi, qlo, out);
  k_colnorm<<<dim3(K_ / 256), dim3(256), 0, stream>>>(queue, invc);
  k_transpose_split<<<dim3(K_ / 64, C_ / 64), dim3(256), 0, stream>>>(queue, invc, Qhi, Qlo);
  k_gemm<<<dim3(K_ / 512, B_ / 16), dim3(256), 0, stream>>>(qhi, qlo, Qhi, Qlo, out);
  k_topk<<<dim3(B_), dim3(256), 0, stream>>>(out, ih);
  k_synth<<<dim3(B_), dim3(256), 0, stream>>>(qn, Qhi, Qlo, ih, u1, u2, u3, noise4,
                                              idx1, idx2, idx3a, idx3b, idx4, idx5, idx6, out);
}
